// TransformerDecoder_19447611916537
// MI455X (gfx1250) — matrix operand order repaired and checked by reading, not compiled, not run on hardware
//
#include <hip/hip_runtime.h>
#include <math.h>

// ---------------------------------------------------------------------------
// Types for CDNA5 WMMA
// ---------------------------------------------------------------------------
typedef __attribute__((ext_vector_type(16))) __bf16 v16bf;
typedef __attribute__((ext_vector_type(8)))  float  v8f;

#define LDK 40  // LDS row stride in ushorts (80B: 16B-aligned, padded vs banks)

__device__ __forceinline__ unsigned short f2bf(float f) {
    unsigned u = __float_as_uint(f);
    u += 0x7FFFu + ((u >> 16) & 1u);   // round-to-nearest-even
    return (unsigned short)(u >> 16);
}

__device__ __forceinline__ void pack_store4(unsigned short* dst, float4 f) {
    ushort4 p;
    p.x = f2bf(f.x); p.y = f2bf(f.y); p.z = f2bf(f.z); p.w = f2bf(f.w);
    *reinterpret_cast<ushort4*>(dst) = p;   // 8B-aligned: kk%4==0, row=80B
}

union FragU { uint4 q[2]; v16bf v; };

// A fragment (16x32 bf16, rows=M): lanes 0-15 hold K 0..7 & 16..23,
// lanes 16-31 hold K 8..15 & 24..31 (ISA 7.12.2, 16-bit A layout).
__device__ __forceinline__ v16bf load_frag_a(const unsigned short* base, int lane) {
    int r  = lane & 15;
    int hi = lane >> 4;
    const unsigned short* p = base + r * LDK + hi * 8;
    FragU u;
    u.q[0] = *reinterpret_cast<const uint4*>(p);        // K = kb .. kb+7
    u.q[1] = *reinterpret_cast<const uint4*>(p + 16);   // K = kb+16 .. kb+23
    return u.v;
}

// B fragment (32x16 bf16, cols=N, stored in LDS as [n][k]): lanes 0-15 hold
// K 0..7 & 16..23 for col n=lane, lanes 16-31 hold K 8..15 & 24..31 (matches A).
__device__ __forceinline__ v16bf load_frag_b(const unsigned short* base, int lane) {
    int r  = lane & 15;
    int hi = lane >> 4;
    const unsigned short* p = base + r * LDK + hi * 8;
    FragU u;
    u.q[0] = *reinterpret_cast<const uint4*>(p);        // K = kb .. kb+7
    u.q[1] = *reinterpret_cast<const uint4*>(p + 16);   // K = kb+16 .. kb+23
    return u.v;
}

// ---------------------------------------------------------------------------
// Universal WMMA GEMM:  C = act(alpha * A@B + bias) (+ resid)
//   A: (M,K) row-major fp32.   B: (K,N) fp32, row-major if !TRANSB,
//   else stored as (N,K) row-major (i.e. B = Bmem^T) -> used for Q·K^T.
//   Batched via blockIdx.z with element strides sA/sB/sC/sR.
//   Block = 256 thr (8 waves). Tile 128x128, wave 32x64, K-step 32.
//   Double-buffered LDS, register-staged prefetch, float4 global loads.
//   M, N must be multiples of 128; K multiple of 32; lda/ldb multiples of 4
//   (true for all call sites here).
// ---------------------------------------------------------------------------
template<bool TRANSB, bool BIAS, bool RELU, bool RESID>
__global__ __launch_bounds__(256) void gemm_wmma(
    const float* __restrict__ A, const float* __restrict__ Bm,
    const float* __restrict__ bias, const float* __restrict__ resid,
    float* __restrict__ C,
    int M, int N, int K, int lda, int ldb, int ldc,
    long sA, long sB, long sC, long sR, float alpha)
{
    __shared__ alignas(16) unsigned short ldsA[2][128 * LDK];
    __shared__ alignas(16) unsigned short ldsB[2][128 * LDK];

    const int t    = threadIdx.x & 255;   // compiler-known range, no guards
    const int lane = t & 31;
    const int wave = t >> 5;
    const int wm   = wave & 3;            // 4 waves along M (32 rows each)
    const int wn   = wave >> 2;           // 2 waves along N (64 cols each)

    const long zb = blockIdx.z;
    A += zb * sA;  Bm += zb * sB;  C += zb * sC;
    const float* R = RESID ? (resid + zb * sR) : nullptr;

    const int mB = blockIdx.y * 128;
    const int nB = blockIdx.x * 128;

    v8f acc[2][4];
    const v8f vz = {0.f,0.f,0.f,0.f,0.f,0.f,0.f,0.f};
#pragma unroll
    for (int mt = 0; mt < 2; ++mt)
#pragma unroll
        for (int nt = 0; nt < 4; ++nt) acc[mt][nt] = vz;

    float4 ra[4], rb[4];   // register staging for the next K-tile

    // load one K-tile (A: 128x32, B: 32x128) into registers (float4 loads)
    auto load_tiles = [&](int k0) {
#pragma unroll
        for (int it = 0; it < 4; ++it) {
            int i = t + it * 256;                         // 0..1023
            int r  = i >> 3;                              // 0..127
            int kk = (i & 7) * 4;                         // 0,4,..,28
            ra[it] = *reinterpret_cast<const float4*>(
                &A[(long)(mB + r) * lda + (k0 + kk)]);
            if (TRANSB) {
                rb[it] = *reinterpret_cast<const float4*>(
                    &Bm[(long)(nB + r) * ldb + (k0 + kk)]);
            } else {
                int bk = i >> 5;                          // 0..31
                int n4 = (i & 31) * 4;                    // 0,4,..,124
                rb[it] = *reinterpret_cast<const float4*>(
                    &Bm[(long)(k0 + bk) * ldb + (nB + n4)]);
            }
        }
    };
    // convert registers to bf16 and store into LDS buffer `buf`
    auto store_tiles = [&](int buf) {
#pragma unroll
        for (int it = 0; it < 4; ++it) {
            int i = t + it * 256;
            int r  = i >> 3;
            int kk = (i & 7) * 4;
            pack_store4(&ldsA[buf][r * LDK + kk], ra[it]);
            if (TRANSB) {
                pack_store4(&ldsB[buf][r * LDK + kk], rb[it]);
            } else {
                int bk = i >> 5;
                int n4 = (i & 31) * 4;
                ldsB[buf][(n4 + 0) * LDK + bk] = f2bf(rb[it].x);
                ldsB[buf][(n4 + 1) * LDK + bk] = f2bf(rb[it].y);
                ldsB[buf][(n4 + 2) * LDK + bk] = f2bf(rb[it].z);
                ldsB[buf][(n4 + 3) * LDK + bk] = f2bf(rb[it].w);
            }
        }
    };

    const int ktiles = K >> 5;
    load_tiles(0);
    store_tiles(0);
    __syncthreads();

    int buf = 0;
    for (int kt = 0; kt < ktiles; ++kt) {
        const bool has_next = (kt + 1) < ktiles;
        if (has_next) load_tiles((kt + 1) << 5);   // overlap with compute

        v16bf afr[2];
#pragma unroll
        for (int mt = 0; mt < 2; ++mt)
            afr[mt] = load_frag_a(&ldsA[buf][(wm * 32 + mt * 16) * LDK], lane);
#pragma unroll
        for (int nt = 0; nt < 4; ++nt) {
            v16bf bfr = load_frag_b(&ldsB[buf][(wn * 64 + nt * 16) * LDK], lane);
#pragma unroll
            for (int mt = 0; mt < 2; ++mt)
                acc[mt][nt] = __builtin_amdgcn_wmma_f32_16x16x32_bf16(
                    false, afr[mt], false, bfr, (short)0, acc[mt][nt],
                    false, false);
        }

        if (has_next) store_tiles(buf ^ 1);        // other buffer: no hazard
        __syncthreads();
        buf ^= 1;
    }

    // ---- epilogue: alpha, bias, relu, residual ----
#pragma unroll
    for (int mt = 0; mt < 2; ++mt) {
#pragma unroll
        for (int nt = 0; nt < 4; ++nt) {
            int n     = nB + wn * 64 + nt * 16 + (lane & 15);
            int mbase = mB + wm * 32 + mt * 16 + (lane >> 4) * 8;
            float bv = BIAS ? bias[n] : 0.f;
#pragma unroll
            for (int r = 0; r < 8; ++r) {
                float vv = acc[mt][nt][r] * alpha + bv;
                if (RELU) vv = fmaxf(vv, 0.f);
                long idx = (long)(mbase + r) * ldc + n;
                if (RESID) vv += R[idx];
                C[idx] = vv;
            }
        }
    }
}

// ---------------------------------------------------------------------------
// Row softmax over S=512 keys (in place) + transposed copy into aux output.
// grid = (Sq, B), block = 256; causal mask is additive -1e7 (matches ref).
// ---------------------------------------------------------------------------
__global__ __launch_bounds__(256) void softmax_rows(
    float* __restrict__ sc, float* __restrict__ wT, int S, int causal)
{
    const int q = blockIdx.x, b = blockIdx.y;
    float* row = sc + ((long)b * S + q) * S;
    __shared__ float red[256];
    const int tid = threadIdx.x & 255;

    float v0 = row[tid];
    float v1 = row[tid + 256];
    if (causal) {
        if (tid > q)       v0 += -1.0e7f;
        if (tid + 256 > q) v1 += -1.0e7f;
    }
    red[tid] = fmaxf(v0, v1);
    __syncthreads();
    for (int s2 = 128; s2 > 0; s2 >>= 1) {
        if (tid < s2) red[tid] = fmaxf(red[tid], red[tid + s2]);
        __syncthreads();
    }
    float mx = red[0];
    __syncthreads();
    float e0 = __expf(v0 - mx), e1 = __expf(v1 - mx);
    red[tid] = e0 + e1;
    __syncthreads();
    for (int s2 = 128; s2 > 0; s2 >>= 1) {
        if (tid < s2) red[tid] += red[tid + s2];
        __syncthreads();
    }
    float inv = 1.0f / red[0];
    e0 *= inv; e1 *= inv;
    row[tid]       = e0;
    row[tid + 256] = e1;
    // transposed aux output (B, Sk, Sq)
    wT[((long)b * S + tid)       * S + q] = e0;
    wT[((long)b * S + tid + 256) * S + q] = e1;
}

// ---------------------------------------------------------------------------
// Embedding gather + sinusoidal positional encoding.
// ---------------------------------------------------------------------------
__global__ __launch_bounds__(256) void embed_pe(
    const int* __restrict__ inp, const float* __restrict__ emb,
    float* __restrict__ ctx, int S, int H)
{
    const long idx = blockIdx.x;               // over B*S
    const int tok  = inp[idx];
    const int s    = (int)(idx % S);
    const float ln10k = 9.210340371976184f;    // ln(10000)
    for (int h = threadIdx.x; h < H; h += 256) {
        int dim = h >> 1;
        float arg = (float)s * __expf(-(2.0f * (float)dim / (float)H) * ln10k);
        float pe  = (h & 1) ? __cosf(arg) : __sinf(arg);
        ctx[idx * H + h] = emb[(long)tok * H + h] + pe;
    }
}

// ---------------------------------------------------------------------------
// Orchestration
// ---------------------------------------------------------------------------
extern "C" void kernel_launch(void* const* d_in, const int* in_sizes, int n_in,
                              void* d_out, int out_size, void* d_ws, size_t ws_size,
                              hipStream_t stream)
{
    constexpr int B = 8, S = 512, H = 1024, V = 32000, L = 6;
    constexpr long HH = (long)H * H;
    constexpr long MH = (long)B * S * H;   // activation elements (4096 x 1024)
    constexpr long SS = (long)B * S * S;   // score elements per layer
    const int   M     = B * S;             // 4096
    const float scale = 0.03125f;          // 1/sqrt(1024)

    if (n_in < 20) return;
    const int*   inputs = (const int*)  d_in[0];
    const float* ann    = (const float*)d_in[1];
    // d_in[2] = hidden_init (unused by the reference)
    const float* emb    = (const float*)d_in[3];
    const float* Wq_s = (const float*)d_in[4],  *bq_s = (const float*)d_in[5];
    const float* Wk_s = (const float*)d_in[6],  *bk_s = (const float*)d_in[7];
    const float* Wv_s = (const float*)d_in[8],  *bv_s = (const float*)d_in[9];
    const float* Wq_c = (const float*)d_in[10], *bq_c = (const float*)d_in[11];
    const float* Wk_c = (const float*)d_in[12], *bk_c = (const float*)d_in[13];
    const float* Wv_c = (const float*)d_in[14], *bv_c = (const float*)d_in[15];
    const float* Wm   = (const float*)d_in[16], *bm   = (const float*)d_in[17];
    const float* Wout = (const float*)d_in[18], *bout = (const float*)d_in[19];

    float* out     = (float*)d_out;                     // (B, S, V)
    float* enc_ws  = out + (long)B * S * V;             // (L, B, Sk, Sq)
    float* self_ws = enc_ws + (long)L * B * S * S;      // (L, B, Sk, Sq)

    // workspace carve-out (fp32): 5*MH + SS = ~88 MB
    float* ws    = (float*)d_ws;
    float* ctx   = ws;
    float* resid = ctx   + MH;
    float* q     = resid + MH;
    float* k     = q     + MH;
    float* v     = k     + MH;
    float* sc    = v     + MH;

    const dim3 blk(256);
    auto grid = [](int m, int n, int z) { return dim3(n / 128, m / 128, z); };

    // contexts = embedding[inputs] + PE
    embed_pe<<<dim3(B * S), blk, 0, stream>>>(inputs, emb, ctx, S, H);

    for (int i = 0; i < L; ++i) {
        // ---------------- self attention (causal) ----------------
        gemm_wmma<false,true,false,false><<<grid(M,H,1), blk, 0, stream>>>(
            ctx, Wq_s + i*HH, bq_s + (long)i*H, nullptr, q,
            M, H, H, H, H, H, 0,0,0,0, 1.0f);
        gemm_wmma<false,true,false,false><<<grid(M,H,1), blk, 0, stream>>>(
            ctx, Wk_s + i*HH, bk_s + (long)i*H, nullptr, k,
            M, H, H, H, H, H, 0,0,0,0, 1.0f);
        gemm_wmma<false,true,false,false><<<grid(M,H,1), blk, 0, stream>>>(
            ctx, Wv_s + i*HH, bv_s + (long)i*H, nullptr, v,
            M, H, H, H, H, H, 0,0,0,0, 1.0f);
        // scores[b] = q[b] @ k[b]^T * scale           (batched, TRANS_B)
        gemm_wmma<true,false,false,false><<<grid(S,S,B), blk, 0, stream>>>(
            q, k, nullptr, nullptr, sc,
            S, S, H, H, H, S, (long)S*H, (long)S*H, (long)S*S, 0, scale);
        softmax_rows<<<dim3(S,B), blk, 0, stream>>>(
            sc, self_ws + (long)i*SS, S, 1);
        // resid = ctx + w @ v                          (batched, +residual)
        gemm_wmma<false,false,false,true><<<grid(S,H,B), blk, 0, stream>>>(
            sc, v, nullptr, ctx, resid,
            S, H, S, S, H, H, (long)S*S, (long)S*H, (long)S*H, (long)S*H, 1.0f);

        // ---------------- cross attention ----------------
        gemm_wmma<false,true,false,false><<<grid(M,H,1), blk, 0, stream>>>(
            resid, Wq_c + i*HH, bq_c + (long)i*H, nullptr, q,
            M, H, H, H, H, H, 0,0,0,0, 1.0f);
        gemm_wmma<false,true,false,false><<<grid(M,H,1), blk, 0, stream>>>(
            ann, Wk_c + i*HH, bk_c + (long)i*H, nullptr, k,
            M, H, H, H, H, H, 0,0,0,0, 1.0f);
        gemm_wmma<false,true,false,false><<<grid(M,H,1), blk, 0, stream>>>(
            ann, Wv_c + i*HH, bv_c + (long)i*H, nullptr, v,
            M, H, H, H, H, H, 0,0,0,0, 1.0f);
        gemm_wmma<true,false,false,false><<<grid(S,S,B), blk, 0, stream>>>(
            q, k, nullptr, nullptr, sc,
            S, S, H, H, H, S, (long)S*H, (long)S*H, (long)S*S, 0, scale);
        softmax_rows<<<dim3(S,B), blk, 0, stream>>>(
            sc, enc_ws + (long)i*SS, S, 0);
        // resid = resid + w @ v  (in-place residual: 1 read + 1 write / elem)
        gemm_wmma<false,false,false,true><<<grid(S,H,B), blk, 0, stream>>>(
            sc, v, nullptr, resid, resid,
            S, H, S, S, H, H, (long)S*S, (long)S*H, (long)S*H, (long)S*H, 1.0f);

        // ---------------- MLP: ctx = resid + relu(resid @ Wm + bm) ----------
        gemm_wmma<false,true,true,true><<<grid(M,H,1), blk, 0, stream>>>(
            resid, Wm + i*HH, bm + (long)i*H, resid, ctx,
            M, H, H, H, H, H, 0,0,0,0, 1.0f);
    }

    // output = ctx @ Wout + bout   (4096 x 32000, K=1024)
    gemm_wmma<false,true,false,false><<<grid(M,V,1), blk, 0, stream>>>(
        ctx, Wout, bout, nullptr, out,
        M, V, H, H, V, V, 0,0,0,0, 1.0f);
}
